// DiceLoss_37675453120526
// MI455X (gfx1250) — 8-bit matrix operand order repaired and checked by reading, compiled, not run on hardware
//
#include <hip/hip_runtime.h>


#define PLANE   (512 * 512)
#define NPIX    (16 * PLANE)
#define NCHUNK  (NPIX / 4)
#define BLOCKS  2048
#define THREADS 256
#define EPSF    1e-6f

typedef int v8i __attribute__((ext_vector_type(8)));

__global__ void dice_zero_ws(int* ws) {
    if (threadIdx.x < 12) ws[threadIdx.x] = 0;
}

__device__ __forceinline__ int argmax4(float a, float b, float c, float d) {
    float m = a; int p = 0;
    if (b > m) { m = b; p = 1; }
    if (c > m) { m = c; p = 2; }
    if (d > m) {        p = 3; }
    return p;
}

__device__ __forceinline__ unsigned int pack_oh(int a, int b, int c, int d, int m) {
    return (unsigned int)(a == m) | ((unsigned int)(b == m) << 8) |
           ((unsigned int)(c == m) << 16) | ((unsigned int)(d == m) << 24);
}

__global__ __launch_bounds__(THREADS)
void dice_count(const float* __restrict__ inp, const int* __restrict__ tgt,
                int* __restrict__ ws) {
    __shared__ unsigned int img[8 * 1024];
    __shared__ int cnt[12];

    const int tid = threadIdx.x;
    for (int i = tid; i < 8 * 1024; i += THREADS) img[i] = 0u;
    if (tid < 12) cnt[tid] = 0;
    __syncthreads();

    const int wave = tid >> 5;
    const int lane = tid & 31;
    const int rnd  = lane >> 4;
    const int g    = lane & 15;

    const int va = (g & 1) + ((g >> 2) << 1);
    const int ha = (g >> 1) & 1;
    const int vb = (g & 1) + ((g >> 2) << 1);
    const int hb = (g >> 1) & 1;

    unsigned int* __restrict__ Aimg = &img[wave * 1024 + rnd * 512];
    unsigned int* __restrict__ Bimg = Aimg + 256;
    const unsigned int* __restrict__ wbase = &img[wave * 1024];

    const int aslot = va * 32 + ha * 16;
    const int bslot = vb * 32 + hb * 16;

    v8i acc = {};

    const int T = BLOCKS * THREADS;
    for (int chunk = blockIdx.x * THREADS + tid; chunk < NCHUNK; chunk += T) {
        const int p0 = chunk << 2;
        const int b  = p0 >> 18;
        const int s  = p0 & (PLANE - 1);
        const size_t ibase = ((size_t)(b * 4) << 18) + (size_t)s;

        const float4 c0 = *(const float4*)(inp + ibase);
        const float4 c1 = *(const float4*)(inp + ibase + PLANE);
        const float4 c2 = *(const float4*)(inp + ibase + 2 * PLANE);
        const float4 c3 = *(const float4*)(inp + ibase + 3 * PLANE);
        const int4   t4 = *(const int4*)(tgt + p0);

        const int pr0 = argmax4(c0.x, c1.x, c2.x, c3.x);
        const int pr1 = argmax4(c0.y, c1.y, c2.y, c3.y);
        const int pr2 = argmax4(c0.z, c1.z, c2.z, c3.z);
        const int pr3 = argmax4(c0.w, c1.w, c2.w, c3.w);

        #pragma unroll
        for (int m = 0; m < 4; ++m) {
            Aimg[aslot + m] = pack_oh(pr0, pr1, pr2, pr3, m);
            Bimg[bslot + m] = pack_oh(t4.x, t4.y, t4.z, t4.w, m);
        }
        Aimg[aslot + 4] = 0x01010101u;
        Bimg[bslot + 4] = 0x01010101u;

        asm volatile("s_wait_dscnt 0x0" ::: "memory");

        #pragma unroll
        for (int r = 0; r < 2; ++r) {
            const unsigned int* __restrict__ Ar = wbase + r * 512;
            const unsigned int* __restrict__ Br = Ar + 256;
            v8i a, bm;
            #pragma unroll
            for (int v = 0; v < 8; ++v) {
                a[v]  = (int)Ar[v * 32 + lane];
                bm[v] = (int)Br[v * 32 + lane];
            }
            acc = __builtin_amdgcn_wmma_i32_16x16x64_iu8(
                      false, a, false, bm, acc, false, false);
        }
        asm volatile("" ::: "memory");
    }

    #pragma unroll
    for (int c = 0; c < 4; ++c) {
        if (lane == c) {
            atomicAdd(&cnt[c], acc[c]);
            atomicAdd(&cnt[8 + c], acc[4]);
        }
    }
    if (lane == 4) {
        #pragma unroll
        for (int c = 0; c < 4; ++c) atomicAdd(&cnt[4 + c], acc[c]);
    }
    __syncthreads();
    if (tid < 12) atomicAdd(&ws[tid], cnt[tid]);
}

__global__ void dice_finalize(const int* __restrict__ ws, float* __restrict__ out) {
    if (threadIdx.x == 0 && blockIdx.x == 0) {
        float loss = 0.0f;
        #pragma unroll
        for (int c = 0; c < 4; ++c) {
            const float inter = (float)ws[c];
            const float uni   = (float)ws[4 + c] + (float)ws[8 + c];
            const float dice  = (2.0f * inter + EPSF) / (uni + EPSF);
            loss += 1.0f - dice;
        }
        out[0] = loss * 0.25f;
    }
}

extern "C" void kernel_launch(void* const* d_in, const int* in_sizes, int n_in,
                              void* d_out, int out_size, void* d_ws, size_t ws_size,
                              hipStream_t stream) {
    const float* inp = (const float*)d_in[0];
    const int*   tgt = (const int*)d_in[1];
    int*   ws  = (int*)d_ws;
    float* out = (float*)d_out;

    dice_zero_ws<<<1, 32, 0, stream>>>(ws);
    dice_count<<<BLOCKS, THREADS, 0, stream>>>(inp, tgt, ws);
    dice_finalize<<<1, 32, 0, stream>>>(ws, out);
}
